// PointTransformerLayer_19920058319196
// MI455X (gfx1250) — matrix operand order repaired and checked by reading, not compiled, not run on hardware
//
#include <hip/hip_runtime.h>
#include <hip/hip_bf16.h>

typedef __attribute__((ext_vector_type(16))) _Float16 v16h;
typedef __attribute__((ext_vector_type(8)))  _Float16 v8h;
typedef __attribute__((ext_vector_type(8)))  float    v8f;

#define NPTS   4096
#define BATCH  8
#define KNN    16
#define CH     64
#define NTOT   (BATCH * NPTS)          // 32768
#define NEDGE  (NTOT * KNN)            // 524288
#define BN_EPS 1e-5f

// ---------------------------------------------------------------------------
// WMMA helper (CDNA5 v_wmma_f32_16x16x32_f16, wave32)
// ---------------------------------------------------------------------------
__device__ __forceinline__ v8f wmma_f16(v16h a, v16h b, v8f c) {
    return __builtin_amdgcn_wmma_f32_16x16x32_f16(false, a, false, b, (short)0, c, false, false);
}

// A fragment from an f16 row: two contiguous 16B chunks per lane.
// element e -> k = kbase + 8*half + (e<8 ? e : e+8)
__device__ __forceinline__ v16h load_a_frag(const _Float16* __restrict__ row,
                                            int kbase, int half) {
    const v8h* p0 = (const v8h*)(row + kbase + 8 * half);
    const v8h* p1 = (const v8h*)(row + kbase + 8 * half + 16);
    v8h lo = *p0, hi = *p1;
    return __builtin_shufflevector(lo, hi, 0, 1, 2, 3, 4, 5, 6, 7,
                                   8, 9, 10, 11, 12, 13, 14, 15);
}

__device__ __forceinline__ v16h pack_frag(v8h lo, v8h hi) {
    return __builtin_shufflevector(lo, hi, 0, 1, 2, 3, 4, 5, 6, 7,
                                   8, 9, 10, 11, 12, 13, 14, 15);
}

// ---------------------------------------------------------------------------
// kNN: one cloud staged in LDS, one center per lane, top-16 insertion sort.
// ---------------------------------------------------------------------------
__global__ void knn_kernel(const float* __restrict__ pos, int* __restrict__ nbr) {
    __shared__ float sx[NPTS], sy[NPTS], sz[NPTS];
    int b   = blockIdx.x / (NPTS / 256);
    int blk = blockIdx.x % (NPTS / 256);
    const float* p = pos + (size_t)b * NPTS * 3;
    for (int j = threadIdx.x; j < NPTS; j += 256) {
        sx[j] = p[j * 3 + 0]; sy[j] = p[j * 3 + 1]; sz[j] = p[j * 3 + 2];
    }
    __syncthreads();
    int ci = blk * 256 + threadIdx.x;
    float px = sx[ci], py = sy[ci], pz = sz[ci];
    float bd[KNN]; int bi[KNN];
#pragma unroll
    for (int t = 0; t < KNN; ++t) { bd[t] = 3.4e38f; bi[t] = 0; }
    for (int j = 0; j < NPTS; ++j) {
        float dx = px - sx[j], dy = py - sy[j], dz = pz - sz[j];
        float d = dx * dx + dy * dy + dz * dz;
        if (d < bd[KNN - 1]) {
            int ins = KNN - 1;
#pragma unroll
            for (int t = KNN - 1; t > 0; --t) {
                if (d < bd[t - 1]) { bd[t] = bd[t - 1]; bi[t] = bi[t - 1]; ins = t - 1; }
            }
            bd[ins] = d; bi[ins] = j;
        }
    }
    int gi = b * NPTS + ci;
#pragma unroll
    for (int t = 0; t < KNN; ++t) nbr[gi * KNN + t] = b * NPTS + bi[t];
}

// ---------------------------------------------------------------------------
// Generic 64-wide WMMA GEMM: out[M,64] = actA(M,64) @ W[64,64] + bias
// A sources are f16. MODE 0: plain; MODE 1: relu(bn(A)); MODE 2: q-k+rho.
// Optional per-channel sum/sumsq epilogue (BatchNorm training stats),
// taken from the f32 accumulators.
// ---------------------------------------------------------------------------
template<int MODE, typename OutT>
__launch_bounds__(256, 1)
__global__ void gemm64_kernel(const _Float16* __restrict__ A, const float* __restrict__ W,
                              const float* __restrict__ bias, OutT* __restrict__ out, int M,
                              const float* __restrict__ bn_mean, const float* __restrict__ bn_rstd,
                              const float* __restrict__ bn_g, const float* __restrict__ bn_beta,
                              const _Float16* __restrict__ qv, const _Float16* __restrict__ kv,
                              const _Float16* __restrict__ rho, const int* __restrict__ nbr,
                              float* __restrict__ stat_sum, float* __restrict__ stat_ssq) {
    __shared__ float s_bn[4][CH];
    if (MODE == 1) {
        for (int t = threadIdx.x; t < CH; t += blockDim.x) {
            s_bn[0][t] = bn_mean[t]; s_bn[1][t] = bn_rstd[t];
            s_bn[2][t] = bn_g[t];    s_bn[3][t] = bn_beta[t];
        }
        __syncthreads();
    }
    int lane = threadIdx.x & 31;
    int half = lane >> 4;
    int n    = lane & 15;           // also A-row m within the tile
    int wave = blockIdx.x * (blockDim.x >> 5) + (threadIdx.x >> 5);
    int nwav = gridDim.x * (blockDim.x >> 5);

    // B fragments: 2 K-steps x 4 N-tiles, built once per wave (f32 -> f16)
    v16h bfrag[2][4];
#pragma unroll
    for (int ks = 0; ks < 2; ++ks) {
#pragma unroll
        for (int nt = 0; nt < 4; ++nt) {
            int koff = ks * 32 + 8 * half;
            int col  = nt * 16 + n;
            v16h bf;
#pragma unroll
            for (int e = 0; e < 16; ++e) bf[e] = (_Float16)W[(koff + ((e < 8) ? e : e + 8)) * CH + col];
            bfrag[ks][nt] = bf;
        }
    }
    float bsv[4];
#pragma unroll
    for (int nt = 0; nt < 4; ++nt) bsv[nt] = bias[nt * 16 + n];

    float lsum[4] = {0.f, 0.f, 0.f, 0.f};
    float lssq[4] = {0.f, 0.f, 0.f, 0.f};

    int mtiles = M >> 4;
    for (int tile = wave; tile < mtiles; tile += nwav) {
        v16h af0, af1;
        if (MODE == 0) {
            const _Float16* arow = A + (size_t)(tile * 16 + n) * CH;
            af0 = load_a_frag(arow, 0, half);
            af1 = load_a_frag(arow, 32, half);
        } else if (MODE == 1) {
            const _Float16* arow = A + (size_t)(tile * 16 + n) * CH;
            af0 = load_a_frag(arow, 0, half);
            af1 = load_a_frag(arow, 32, half);
#pragma unroll
            for (int e = 0; e < 16; ++e) {
                int k0 = 8 * half + ((e < 8) ? e : e + 8);
                float v0 = ((float)af0[e] - s_bn[0][k0]) * s_bn[1][k0] * s_bn[2][k0] + s_bn[3][k0];
                af0[e] = (_Float16)fmaxf(v0, 0.f);
                int k1 = k0 + 32;
                float v1 = ((float)af1[e] - s_bn[0][k1]) * s_bn[1][k1] * s_bn[2][k1] + s_bn[3][k1];
                af1[e] = (_Float16)fmaxf(v1, 0.f);
            }
        } else {
            int e = tile * 16 + n;              // KNN==16: tile == center index
            int j = nbr[e];
            const _Float16* qr = qv  + (size_t)tile * CH;
            const _Float16* kr = kv  + (size_t)j    * CH;
            const _Float16* rr = rho + (size_t)e    * CH;
#pragma unroll
            for (int ks = 0; ks < 2; ++ks) {
                int o0 = ks * 32 + 8 * half;
                v8h lo = *(const v8h*)(qr + o0) - *(const v8h*)(kr + o0) + *(const v8h*)(rr + o0);
                v8h hi = *(const v8h*)(qr + o0 + 16) - *(const v8h*)(kr + o0 + 16) + *(const v8h*)(rr + o0 + 16);
                if (ks == 0) af0 = pack_frag(lo, hi); else af1 = pack_frag(lo, hi);
            }
        }

        v8f acc[4];
#pragma unroll
        for (int nt = 0; nt < 4; ++nt) {
            v8f c;
#pragma unroll
            for (int r = 0; r < 8; ++r) c[r] = bsv[nt];
            c = wmma_f16(af0, bfrag[0][nt], c);
            c = wmma_f16(af1, bfrag[1][nt], c);
            acc[nt] = c;
        }
#pragma unroll
        for (int nt = 0; nt < 4; ++nt) {
#pragma unroll
            for (int r = 0; r < 8; ++r) {
                float v = acc[nt][r];
                out[(size_t)(tile * 16 + r + 8 * half) * CH + nt * 16 + n] = (OutT)v;
                lsum[nt] += v;
                lssq[nt] += v * v;
            }
        }
    }
    if (stat_sum != nullptr) {
#pragma unroll
        for (int nt = 0; nt < 4; ++nt) {
            atomicAdd(&stat_sum[nt * 16 + n], lsum[nt]);
            atomicAdd(&stat_ssq[nt * 16 + n], lssq[nt]);
        }
    }
}

// ---------------------------------------------------------------------------
// f32 -> f16 conversion (for x)
// ---------------------------------------------------------------------------
__global__ void cvt_f16_kernel(const float* __restrict__ in, _Float16* __restrict__ out, int n) {
    int t = blockIdx.x * blockDim.x + threadIdx.x;
    if (t < n) out[t] = (_Float16)in[t];
}

// ---------------------------------------------------------------------------
// rho layer-1: e1[e,c] = delta_p(e) . W1[:,c] + b1[c]; fused BN stats (f32).
// ---------------------------------------------------------------------------
__global__ void rho1_kernel(const float* __restrict__ pos, const int* __restrict__ nbr,
                            const float* __restrict__ W1, const float* __restrict__ b1,
                            _Float16* __restrict__ e1,
                            float* __restrict__ ssum, float* __restrict__ sssq) {
    __shared__ float s_sum[CH], s_ssq[CH];
    if (threadIdx.x < CH) { s_sum[threadIdx.x] = 0.f; s_ssq[threadIdx.x] = 0.f; }
    __syncthreads();
    int c = threadIdx.x & (CH - 1);
    int e = blockIdx.x * 4 + (threadIdx.x >> 6);
    int i = e >> 4;
    int j = nbr[e];
    float dx = pos[(size_t)i * 3 + 0] - pos[(size_t)j * 3 + 0];
    float dy = pos[(size_t)i * 3 + 1] - pos[(size_t)j * 3 + 1];
    float dz = pos[(size_t)i * 3 + 2] - pos[(size_t)j * 3 + 2];
    float v = dx * W1[0 * CH + c] + dy * W1[1 * CH + c] + dz * W1[2 * CH + c] + b1[c];
    e1[(size_t)e * CH + c] = (_Float16)v;
    atomicAdd(&s_sum[c], v);
    atomicAdd(&s_ssq[c], v * v);
    __syncthreads();
    if (threadIdx.x < CH) {
        atomicAdd(&ssum[threadIdx.x], s_sum[threadIdx.x]);
        atomicAdd(&sssq[threadIdx.x], s_ssq[threadIdx.x]);
    }
}

__global__ void bn_finalize_kernel(const float* __restrict__ ssum, const float* __restrict__ sssq,
                                   float* __restrict__ mean, float* __restrict__ rstd, float cnt) {
    int c = threadIdx.x;
    if (c < CH) {
        float mu  = ssum[c] / cnt;
        float var = sssq[c] / cnt - mu * mu;
        mean[c] = mu;
        rstd[c] = rsqrtf(var + BN_EPS);
    }
}

// ---------------------------------------------------------------------------
// Softmax over K per (center, channel) + weighted aggregation of (v + rho).
// ---------------------------------------------------------------------------
__global__ void softmax_agg_kernel(const _Float16* __restrict__ attn, const _Float16* __restrict__ vv,
                                   const _Float16* __restrict__ rho, const int* __restrict__ nbr,
                                   _Float16* __restrict__ agg) {
    __shared__ int s_nbr[4][KNN];
    int li = threadIdx.x >> 6;
    int c  = threadIdx.x & (CH - 1);
    int i  = blockIdx.x * 4 + li;
    if (threadIdx.x < 4 * KNN) {
        int lc = threadIdx.x >> 4, kk = threadIdx.x & 15;
        s_nbr[lc][kk] = nbr[(size_t)(blockIdx.x * 4 + lc) * KNN + kk];
    }
    __syncthreads();
    const _Float16* arow = attn + (size_t)i * KNN * CH + c;
    float av[KNN];
    float mx = -3.4e38f;
#pragma unroll
    for (int k = 0; k < KNN; ++k) { av[k] = (float)arow[k * CH]; mx = fmaxf(mx, av[k]); }
    float s = 0.f;
#pragma unroll
    for (int k = 0; k < KNN; ++k) { av[k] = __expf(av[k] - mx); s += av[k]; }
    float inv = 1.f / s;
    const _Float16* rrow = rho + (size_t)i * KNN * CH + c;
    float acc = 0.f;
#pragma unroll
    for (int k = 0; k < KNN; ++k) {
        int j = s_nbr[li][k];
        acc += av[k] * ((float)vv[(size_t)j * CH + c] + (float)rrow[k * CH]);
    }
    agg[(size_t)i * CH + c] = (_Float16)(acc * inv);
}

__global__ void zero_kernel(float* __restrict__ p, int n) {
    int t = blockIdx.x * blockDim.x + threadIdx.x;
    if (t < n) p[t] = 0.f;
}

// ---------------------------------------------------------------------------
extern "C" void kernel_launch(void* const* d_in, const int* in_sizes, int n_in,
                              void* d_out, int out_size, void* d_ws, size_t ws_size,
                              hipStream_t stream) {
    const float* x    = (const float*)d_in[0];
    const float* pos  = (const float*)d_in[1];
    /* d_in[2] = batch (int64), layout is known: unused */
    const float* Wq   = (const float*)d_in[3];  const float* bq    = (const float*)d_in[4];
    const float* Wk   = (const float*)d_in[5];  const float* bk    = (const float*)d_in[6];
    const float* Wv   = (const float*)d_in[7];  const float* bv    = (const float*)d_in[8];
    const float* rW1  = (const float*)d_in[9];  const float* rb1   = (const float*)d_in[10];
    const float* rg   = (const float*)d_in[11]; const float* rbeta = (const float*)d_in[12];
    const float* rW2  = (const float*)d_in[13]; const float* rb2   = (const float*)d_in[14];
    const float* gW1  = (const float*)d_in[15]; const float* gb1   = (const float*)d_in[16];
    const float* gg   = (const float*)d_in[17]; const float* gbeta = (const float*)d_in[18];
    const float* gW2  = (const float*)d_in[19]; const float* gb2   = (const float*)d_in[20];
    const float* Wo   = (const float*)d_in[21]; const float* bo    = (const float*)d_in[22];
    float* outp = (float*)d_out;

    char* ws = (char*)d_ws;
    auto walloc = [&](size_t bytes) {
        char* p = ws; ws += (bytes + 255) & ~(size_t)255; return p;
    };
    int*      nbr   = (int*)     walloc(sizeof(int)      * (size_t)NEDGE);
    _Float16* xh    = (_Float16*)walloc(sizeof(_Float16) * (size_t)NTOT * CH);
    _Float16* q     = (_Float16*)walloc(sizeof(_Float16) * (size_t)NTOT * CH);
    _Float16* kk    = (_Float16*)walloc(sizeof(_Float16) * (size_t)NTOT * CH);
    _Float16* vv    = (_Float16*)walloc(sizeof(_Float16) * (size_t)NTOT * CH);
    _Float16* buf1  = (_Float16*)walloc(sizeof(_Float16) * (size_t)NEDGE * CH); // e1 -> rho
    _Float16* buf2  = (_Float16*)walloc(sizeof(_Float16) * (size_t)NEDGE * CH); // a1 -> attn
    _Float16* agg   = (_Float16*)walloc(sizeof(_Float16) * (size_t)NTOT * CH);
    float*    stats = (float*)   walloc(sizeof(float) * 8 * CH);
    float* r_sum = stats + 0 * CH; float* r_ssq = stats + 1 * CH;
    float* r_mu  = stats + 2 * CH; float* r_rs  = stats + 3 * CH;
    float* g_sum = stats + 4 * CH; float* g_ssq = stats + 5 * CH;
    float* g_mu  = stats + 6 * CH; float* g_rs  = stats + 7 * CH;

    dim3 blk(256);
    zero_kernel<<<1, 512, 0, stream>>>(stats, 8 * CH);
    knn_kernel<<<BATCH * (NPTS / 256), blk, 0, stream>>>(pos, nbr);
    cvt_f16_kernel<<<(NTOT * CH) / 256, blk, 0, stream>>>(x, xh, NTOT * CH);

    // q, k, v projections (f16 out)
    gemm64_kernel<0, _Float16><<<256, blk, 0, stream>>>(xh, Wq, bq, q, NTOT,
        nullptr, nullptr, nullptr, nullptr, nullptr, nullptr, nullptr, nullptr, nullptr, nullptr);
    gemm64_kernel<0, _Float16><<<256, blk, 0, stream>>>(xh, Wk, bk, kk, NTOT,
        nullptr, nullptr, nullptr, nullptr, nullptr, nullptr, nullptr, nullptr, nullptr, nullptr);
    gemm64_kernel<0, _Float16><<<256, blk, 0, stream>>>(xh, Wv, bv, vv, NTOT,
        nullptr, nullptr, nullptr, nullptr, nullptr, nullptr, nullptr, nullptr, nullptr, nullptr);

    // rho MLP: layer1 (+stats) -> BN finalize -> fused BN+ReLU+GEMM (in place)
    rho1_kernel<<<NEDGE / 4, blk, 0, stream>>>(pos, nbr, rW1, rb1, buf1, r_sum, r_ssq);
    bn_finalize_kernel<<<1, CH, 0, stream>>>(r_sum, r_ssq, r_mu, r_rs, (float)NEDGE);
    gemm64_kernel<1, _Float16><<<1024, blk, 0, stream>>>(buf1, rW2, rb2, buf1, NEDGE,
        r_mu, r_rs, rg, rbeta, nullptr, nullptr, nullptr, nullptr, nullptr, nullptr);

    // gamma MLP: fused (q-k+rho) construction + GEMM + stats -> BN -> GEMM
    gemm64_kernel<2, _Float16><<<1024, blk, 0, stream>>>(nullptr, gW1, gb1, buf2, NEDGE,
        nullptr, nullptr, nullptr, nullptr, q, kk, buf1, nbr, g_sum, g_ssq);
    bn_finalize_kernel<<<1, CH, 0, stream>>>(g_sum, g_ssq, g_mu, g_rs, (float)NEDGE);
    gemm64_kernel<1, _Float16><<<1024, blk, 0, stream>>>(buf2, gW2, gb2, buf2, NEDGE,
        g_mu, g_rs, gg, gbeta, nullptr, nullptr, nullptr, nullptr, nullptr, nullptr);

    // softmax over K + aggregation, then output projection into d_out (f32)
    softmax_agg_kernel<<<NTOT / 4, blk, 0, stream>>>(buf2, vv, buf1, nbr, agg);
    gemm64_kernel<0, float><<<256, blk, 0, stream>>>(agg, Wo, bo, outp, NTOT,
        nullptr, nullptr, nullptr, nullptr, nullptr, nullptr, nullptr, nullptr, nullptr, nullptr);
}
